// TwoSimplicialAttention_33517924778614
// MI455X (gfx1250) — matrix operand order repaired and checked by reading, not compiled, not run on hardware
//
#include <hip/hip_runtime.h>
#include <hip/hip_bf16.h>

typedef __attribute__((ext_vector_type(16))) _Float16 v16h;
typedef __attribute__((ext_vector_type(8)))  _Float16 v8h;
typedef __attribute__((ext_vector_type(8)))  float    v8f;

#define N_NODES 4096
#define DEG     32
#define DIM     512
#define HEADS   8
#define HDIM    64
#define LN_EPS  1e-5f

union V16u { v16h v; v8h h[2]; };

// A-fragment (16-bit, 16x32): lane L -> row M=L%16; elems 0..7 = K base..base+7,
// elems 8..15 = K 16+base..16+base+7, base = (L<16 ? 0 : 8). Caller pre-adds base.
static __device__ __forceinline__ v16h loadA_frag(const _Float16* p) {
    V16u u;
    u.h[0] = *(const v8h*)(p);
    u.h[1] = *(const v8h*)(p + 16);
    return u.v;
}
// B-fragment (16-bit, 32x16): lane L -> col N=L%16; elems 0..7 = K base..base+7,
// elems 8..15 = K 16+base..16+base+7, base = (L<16 ? 0 : 8). Caller pre-adds base.
static __device__ __forceinline__ v16h loadB_frag(const _Float16* p) {
    V16u u;
    u.h[0] = *(const v8h*)(p);
    u.h[1] = *(const v8h*)(p + 16);
    return u.v;
}

static __device__ __forceinline__ v8f wmma_f16(v16h a, v16h b, v8f c) {
    return __builtin_amdgcn_wmma_f32_16x16x32_f16(false, a, false, b, (short)0, c,
                                                  false, false);
}

// ---------------- conversion kernels ----------------
__global__ void k_conv_x(const float* __restrict__ x, _Float16* __restrict__ xh, int n) {
    int i = blockIdx.x * 256 + threadIdx.x;
    if (i < n) xh[i] = (_Float16)x[i];
}

// Store weights transposed: WT[p][n*DIM + k] = W_p[k*DIM + n]  (B-fragment friendly)
__global__ void k_conv_wT(const float* __restrict__ w0, const float* __restrict__ w1,
                          const float* __restrict__ w2, const float* __restrict__ w3,
                          const float* __restrict__ w4, const float* __restrict__ w5,
                          _Float16* __restrict__ wt) {
    int p = blockIdx.y;
    const float* w = (p == 0) ? w0 : (p == 1) ? w1 : (p == 2) ? w2
                   : (p == 3) ? w3 : (p == 4) ? w4 : w5;
    int i = blockIdx.x * 256 + threadIdx.x;
    if (i < DIM * DIM) {
        int n = i / DIM, k = i % DIM;
        wt[(size_t)p * DIM * DIM + (size_t)n * DIM + k] = (_Float16)w[(size_t)k * DIM + n];
    }
}

// ---------------- fused 5-way projection GEMM ----------------
// grid = (N/16, 5), block = 256 (8 waves). Wave w computes rows[16] x cols[w*64..w*64+64).
__global__ __launch_bounds__(256) void k_proj(
    const _Float16* __restrict__ xh, const _Float16* __restrict__ wt,
    const float* __restrict__ bq, const float* __restrict__ bk, const float* __restrict__ bv,
    const float* __restrict__ bkp, const float* __restrict__ bvp,
    float* __restrict__ qf, _Float16* __restrict__ kh, _Float16* __restrict__ vh,
    _Float16* __restrict__ kph, _Float16* __restrict__ vph)
{
    int p = blockIdx.y;
    int rowBase = blockIdx.x * 16;
    int wave = threadIdx.x >> 5, lane = threadIdx.x & 31;
    int l16 = lane & 15;
    bool hi = lane >= 16;

    const _Float16* W = wt + (size_t)p * DIM * DIM;
    const float* bias = (p == 0) ? bq : (p == 1) ? bk : (p == 2) ? bv : (p == 3) ? bkp : bvp;

    v8f zero = {0.f, 0.f, 0.f, 0.f, 0.f, 0.f, 0.f, 0.f};
    v8f acc[4] = {zero, zero, zero, zero};

    const _Float16* aBase = xh + (size_t)(rowBase + l16) * DIM + (hi ? 8 : 0);
    for (int kc = 0; kc < 16; ++kc) {
        __builtin_prefetch(aBase + (kc + 1) * 32, 0, 0);  // global_prefetch_b8
        v16h a = loadA_frag(aBase + kc * 32);
        for (int nt = 0; nt < 4; ++nt) {
            int col = wave * 64 + nt * 16 + l16;
            v16h b = loadB_frag(W + (size_t)col * DIM + kc * 32 + (hi ? 8 : 0));
            acc[nt] = wmma_f16(a, b, acc[nt]);
        }
    }
    for (int nt = 0; nt < 4; ++nt) {
        int col = wave * 64 + nt * 16 + l16;
        float bb = bias[col];
        for (int r = 0; r < 8; ++r) {
            int orow = rowBase + (hi ? r + 8 : r);
            float val = acc[nt][r] + bb;
            size_t off = (size_t)orow * DIM + col;
            if      (p == 0) qf[off]  = val;
            else if (p == 1) kh[off]  = (_Float16)val;
            else if (p == 2) vh[off]  = (_Float16)val;
            else if (p == 3) kph[off] = (_Float16)val;
            else             vph[off] = (_Float16)val;
        }
    }
}

// ---------------- triadic attention: one wave32 per (node, head) ----------------
__global__ __launch_bounds__(32) void k_attn(
    const float* __restrict__ qf, const _Float16* __restrict__ kh,
    const _Float16* __restrict__ vh, const _Float16* __restrict__ kph,
    const _Float16* __restrict__ vph, const int* __restrict__ edge,
    _Float16* __restrict__ zh)
{
    __shared__ int sIdx[DEG];
    __shared__ __align__(16) float sQ[HDIM];
    __shared__ __align__(16) _Float16 sQKp[DEG * HDIM];  // [k][d] = Q[d]*K'n[k][d]
    __shared__ __align__(16) _Float16 sVn[DEG * HDIM];   // [j][d]
    __shared__ __align__(16) _Float16 sVpT[HDIM * DEG];  // [d][k] (transposed V'n)
    __shared__ __align__(16) _Float16 sS[DEG * DEG];     // softmax probs [j][k]

    int n = blockIdx.x >> 3, h = blockIdx.x & 7;
    int lane = threadIdx.x;
    int l16 = lane & 15;
    bool hi = lane >= 16;

    int e = edge[n * DEG + lane];
    bool valid = e >= 0;
    unsigned vmask = (unsigned)__ballot(valid);
    int idx = valid ? e : 0;
    sIdx[lane] = idx;
    size_t qoff = (size_t)n * DIM + h * HDIM;
    sQ[lane]      = qf[qoff + lane];
    sQ[lane + 32] = qf[qoff + lane + 32];
    __syncthreads();

    if (vmask == 0u) {  // isolated node: Z = Q
        zh[qoff + lane]      = (_Float16)sQ[lane];
        zh[qoff + lane + 32] = (_Float16)sQ[lane + 32];
        return;
    }

    // Stage: lane j owns neighbor j
    {
        const _Float16* kpRow = kph + (size_t)idx * DIM + h * HDIM;
        const _Float16* vRow  = vh  + (size_t)idx * DIM + h * HDIM;
        const _Float16* vpRow = vph + (size_t)idx * DIM + h * HDIM;
        for (int d = 0; d < HDIM; ++d) {
            sQKp[lane * HDIM + d] = (_Float16)((float)kpRow[d] * sQ[d]);
            sVn [lane * HDIM + d] = vRow[d];
            sVpT[d * DEG + lane]  = vpRow[d];
        }
    }
    __syncthreads();

    // A[j,k] = sum_d Kn[j,d] * QKp[k,d]   (32x32, K=64 -> 2x2x2 WMMA)
    v8f zero = {0.f, 0.f, 0.f, 0.f, 0.f, 0.f, 0.f, 0.f};
    v8f accA[2][2] = {{zero, zero}, {zero, zero}};
    for (int kc = 0; kc < 2; ++kc) {
        v16h aF[2];
        for (int mt = 0; mt < 2; ++mt) {
            int j = mt * 16 + l16;
            const _Float16* ab = kh + (size_t)sIdx[j] * DIM + h * HDIM + kc * 32 + (hi ? 8 : 0);
            aF[mt] = loadA_frag(ab);
        }
        for (int nt = 0; nt < 2; ++nt) {
            v16h bF = loadB_frag(sQKp + (nt * 16 + l16) * HDIM + kc * 32 + (hi ? 8 : 0));
            for (int mt = 0; mt < 2; ++mt)
                accA[mt][nt] = wmma_f16(aF[mt], bF, accA[mt][nt]);
        }
    }

    // scale + pair mask + softmax over all 1024 (j,k) (whole-wave reduction)
    float m = -1e30f;
    for (int mt = 0; mt < 2; ++mt)
        for (int nt = 0; nt < 2; ++nt)
            for (int r = 0; r < 8; ++r) {
                int j = mt * 16 + (hi ? r + 8 : r);
                int k = nt * 16 + l16;
                bool pv = ((vmask >> j) & 1u) && ((vmask >> k) & 1u);
                float a = pv ? accA[mt][nt][r] * 0.125f : -1e9f;
                accA[mt][nt][r] = a;
                m = fmaxf(m, a);
            }
    for (int off = 16; off >= 1; off >>= 1) m = fmaxf(m, __shfl_xor(m, off, 32));
    float s = 0.f;
    for (int mt = 0; mt < 2; ++mt)
        for (int nt = 0; nt < 2; ++nt)
            for (int r = 0; r < 8; ++r) {
                float pexp = __expf(accA[mt][nt][r] - m);
                accA[mt][nt][r] = pexp;
                s += pexp;
            }
    for (int off = 16; off >= 1; off >>= 1) s += __shfl_xor(s, off, 32);
    float inv = 1.f / s;
    for (int mt = 0; mt < 2; ++mt)
        for (int nt = 0; nt < 2; ++nt)
            for (int r = 0; r < 8; ++r) {
                int j = mt * 16 + (hi ? r + 8 : r);
                int k = nt * 16 + l16;
                sS[j * DEG + k] = (_Float16)(accA[mt][nt][r] * inv);
            }
    __syncthreads();

    // T[j,d] = sum_k S[j,k] * V'n[k,d]   (32x64, K=32 -> 2x4 WMMA)
    v8f accT[2][4] = {{zero, zero, zero, zero}, {zero, zero, zero, zero}};
    v16h bT[4];
    for (int nt = 0; nt < 4; ++nt)
        bT[nt] = loadB_frag(sVpT + (nt * 16 + l16) * DEG + (hi ? 8 : 0));
    for (int mt = 0; mt < 2; ++mt) {
        v16h aF = loadA_frag(sS + (mt * 16 + l16) * DEG + (hi ? 8 : 0));
        for (int nt = 0; nt < 4; ++nt)
            accT[mt][nt] = wmma_f16(aF, bT[nt], accT[mt][nt]);
    }

    // Z[d] = sum_j Vn[j,d] * T[j,d]  (elementwise + cross-half shuffle reduce)
    float z[4] = {0.f, 0.f, 0.f, 0.f};
    for (int nt = 0; nt < 4; ++nt) {
        int d = nt * 16 + l16;
        for (int mt = 0; mt < 2; ++mt)
            for (int r = 0; r < 8; ++r) {
                int j = mt * 16 + (hi ? r + 8 : r);
                z[nt] += (float)sVn[j * HDIM + d] * accT[mt][nt][r];
            }
    }
    for (int nt = 0; nt < 4; ++nt) z[nt] += __shfl_xor(z[nt], 16, 32);
    if (!hi)
        for (int nt = 0; nt < 4; ++nt)
            zh[qoff + nt * 16 + l16] = (_Float16)z[nt];
}

// ---------------- output GEMM + bias + residual + LayerNorm ----------------
// grid = N/16, block = 256 (8 waves). Full 16x512 row block per workgroup.
__global__ __launch_bounds__(256) void k_out(
    const _Float16* __restrict__ zh, const _Float16* __restrict__ woT,
    const float* __restrict__ bo, const float* __restrict__ x,
    const float* __restrict__ g, const float* __restrict__ b,
    float* __restrict__ out)
{
    __shared__ float sOut[16 * DIM];  // 32 KB
    int rowBase = blockIdx.x * 16;
    int wave = threadIdx.x >> 5, lane = threadIdx.x & 31;
    int l16 = lane & 15;
    bool hi = lane >= 16;

    v8f zero = {0.f, 0.f, 0.f, 0.f, 0.f, 0.f, 0.f, 0.f};
    v8f acc[4] = {zero, zero, zero, zero};
    const _Float16* aBase = zh + (size_t)(rowBase + l16) * DIM + (hi ? 8 : 0);
    for (int kc = 0; kc < 16; ++kc) {
        __builtin_prefetch(aBase + (kc + 1) * 32, 0, 0);
        v16h a = loadA_frag(aBase + kc * 32);
        for (int nt = 0; nt < 4; ++nt) {
            int col = wave * 64 + nt * 16 + l16;
            v16h bF = loadB_frag(woT + (size_t)col * DIM + kc * 32 + (hi ? 8 : 0));
            acc[nt] = wmma_f16(a, bF, acc[nt]);
        }
    }
    for (int nt = 0; nt < 4; ++nt) {
        int col = wave * 64 + nt * 16 + l16;
        float bb = bo[col];
        for (int r = 0; r < 8; ++r) {
            int row = hi ? r + 8 : r;
            sOut[row * DIM + col] =
                acc[nt][r] + bb + x[(size_t)(rowBase + row) * DIM + col];
        }
    }
    __syncthreads();

    for (int rr = 0; rr < 2; ++rr) {
        int row = wave * 2 + rr;
        float s = 0.f, sq = 0.f;
        for (int c = lane; c < DIM; c += 32) {
            float v = sOut[row * DIM + c];
            s += v; sq += v * v;
        }
        for (int off = 16; off >= 1; off >>= 1) {
            s  += __shfl_xor(s, off, 32);
            sq += __shfl_xor(sq, off, 32);
        }
        float mu = s * (1.f / DIM);
        float var = sq * (1.f / DIM) - mu * mu;
        float rstd = rsqrtf(var + LN_EPS);
        for (int c = lane; c < DIM; c += 32) {
            float v = (sOut[row * DIM + c] - mu) * rstd * g[c] + b[c];
            out[(size_t)(rowBase + row) * DIM + c] = v;
        }
    }
}

// ---------------- host launcher ----------------
extern "C" void kernel_launch(void* const* d_in, const int* in_sizes, int n_in,
                              void* d_out, int out_size, void* d_ws, size_t ws_size,
                              hipStream_t stream) {
    const float* tri   = (const float*)d_in[0];
    const int*   edge  = (const int*)d_in[1];
    const float* w_q   = (const float*)d_in[2];
    const float* b_q   = (const float*)d_in[3];
    const float* w_k   = (const float*)d_in[4];
    const float* b_k   = (const float*)d_in[5];
    const float* w_v   = (const float*)d_in[6];
    const float* b_v   = (const float*)d_in[7];
    const float* w_kp  = (const float*)d_in[8];
    const float* b_kp  = (const float*)d_in[9];
    const float* w_vp  = (const float*)d_in[10];
    const float* b_vp  = (const float*)d_in[11];
    const float* w_o   = (const float*)d_in[12];
    const float* b_o   = (const float*)d_in[13];
    const float* ln_g  = (const float*)d_in[14];
    const float* ln_b  = (const float*)d_in[15];
    float* out = (float*)d_out;

    char* ws = (char*)d_ws;
    size_t off = 0;
    _Float16* Xh  = (_Float16*)(ws + off); off += (size_t)N_NODES * DIM * 2;       // 4 MB
    _Float16* WT  = (_Float16*)(ws + off); off += (size_t)6 * DIM * DIM * 2;       // 3 MB
    float*    Qf  = (float*)   (ws + off); off += (size_t)N_NODES * DIM * 4;       // 8 MB
    _Float16* Kh  = (_Float16*)(ws + off); off += (size_t)N_NODES * DIM * 2;
    _Float16* Vh  = (_Float16*)(ws + off); off += (size_t)N_NODES * DIM * 2;
    _Float16* Kph = (_Float16*)(ws + off); off += (size_t)N_NODES * DIM * 2;
    _Float16* Vph = (_Float16*)(ws + off); off += (size_t)N_NODES * DIM * 2;
    _Float16* Zh  = (_Float16*)(ws + off); off += (size_t)N_NODES * DIM * 2;
    (void)ws_size; (void)n_in; (void)in_sizes; (void)out_size;

    const _Float16* WoT = WT + (size_t)5 * DIM * DIM;

    int nx = N_NODES * DIM;
    k_conv_x<<<(nx + 255) / 256, 256, 0, stream>>>(tri, Xh, nx);
    dim3 gw((DIM * DIM + 255) / 256, 6);
    k_conv_wT<<<gw, 256, 0, stream>>>(w_q, w_k, w_v, w_kp, w_vp, w_o, WT);

    dim3 gp(N_NODES / 16, 5);
    k_proj<<<gp, 256, 0, stream>>>(Xh, WT, b_q, b_k, b_v, b_kp, b_vp,
                                   Qf, Kh, Vh, Kph, Vph);

    k_attn<<<N_NODES * HEADS, 32, 0, stream>>>(Qf, Kh, Vh, Kph, Vph, edge, Zh);

    k_out<<<N_NODES / 16, 256, 0, stream>>>(Zh, WoT, b_o, tri, ln_g, ln_b, out);
}
